// CrossAttention_G_88459146428998
// MI455X (gfx1250) — matrix operand order repaired and checked by reading, not compiled, not run on hardware
//
#include <hip/hip_runtime.h>
#include <hip/hip_bf16.h>

#define BB 16
#define NN 1024
#define DD 768
#define QK_SCALE 0.03608439182435161f   // 768^-0.5

typedef unsigned short u16;
typedef unsigned int   u32;
typedef __attribute__((ext_vector_type(16))) __bf16 v16bf;
typedef __attribute__((ext_vector_type(8)))  float  v8f;

union Frag4 { uint4 q[2]; v16bf v; };   // 32B WMMA operand = two 16B chunks
union AccF  { v8f v; float f[8]; };

__device__ __forceinline__ u16 f2bf(float f) {
  u32 u = __builtin_bit_cast(u32, f);
  u32 r = (u + 0x7FFFu + ((u >> 16) & 1u)) >> 16;   // round-to-nearest-even
  return (u16)r;
}
__device__ __forceinline__ float bflo(u32 d) {
  return __builtin_bit_cast(float, d << 16);
}
__device__ __forceinline__ float bfhi(u32 d) {
  return __builtin_bit_cast(float, d & 0xFFFF0000u);
}
__device__ __forceinline__ u32 pk2(float a, float b) {
  return (u32)f2bf(a) | ((u32)f2bf(b) << 16);
}
__device__ __forceinline__ float sum8(uint4 q) {
  return bflo(q.x) + bfhi(q.x) + bflo(q.y) + bfhi(q.y) +
         bflo(q.z) + bfhi(q.z) + bflo(q.w) + bfhi(q.w);
}
__device__ __forceinline__ v8f wmma_bf16(v16bf a, v16bf b, v8f c) {
  return __builtin_amdgcn_wmma_f32_16x16x32_bf16(false, a, false, b, (short)0, c,
                                                 false, false);
}
// A operand, row-major bf16 row base (per-lane: row m, + hf*16 bytes applied by caller):
// chunk0 at cbyte, chunk1 at cbyte+32   (ISA 7.12.2 16-bit A 16x32 layout)
__device__ __forceinline__ v16bf fragA_row(const char* arow, int cbyte) {
  Frag4 f;
  f.q[0] = *(const uint4*)(arow + cbyte);
  f.q[1] = *(const uint4*)(arow + cbyte + 32);
  return f.v;
}
// B operand from B^T row-major (per-lane: row n, + hf*16 bytes applied by caller):
// chunk0 at cbyte, chunk1 at cbyte+32
__device__ __forceinline__ v16bf fragB_row(const char* brow, int cbyte) {
  Frag4 f;
  f.q[0] = *(const uint4*)(brow + cbyte);
  f.q[1] = *(const uint4*)(brow + cbyte + 32);
  return f.v;
}

// ---------------- W -> bf16 ----------------
__global__ void convw_kernel(const float* __restrict__ W, u16* __restrict__ wb) {
  int i = (blockIdx.x * blockDim.x + threadIdx.x) * 4;   // grid covers D*D exactly
  float4 f = *(const float4*)(W + i);
  u32* dst = (u32*)(wb + i);
  dst[0] = pk2(f.x, f.y);
  dst[1] = pk2(f.z, f.w);
}

// ---------------- y[b,m,e] -> ybT[b,e,m] bf16 (32x32 tiles) ----------------
__global__ void transp_kernel(const float* __restrict__ y, u16* __restrict__ ybT) {
  int bid = blockIdx.x;
  int et = bid % (DD / 32);
  int mt = (bid / (DD / 32)) % (NN / 32);
  int b  = bid / ((DD / 32) * (NN / 32));
  int m0 = mt * 32, e0 = et * 32;
  __shared__ float tile[32][33];
  int tx = threadIdx.x & 31, ty = threadIdx.x >> 5;  // 256 thr: ty 0..7
#pragma unroll
  for (int i = 0; i < 4; ++i) {
    int r = ty + 8 * i;
    tile[r][tx] = y[((size_t)b * NN + m0 + r) * DD + e0 + tx];
  }
  __syncthreads();
#pragma unroll
  for (int i = 0; i < 4; ++i) {
    int r = ty + 8 * i;
    ybT[((size_t)b * DD + e0 + r) * NN + m0 + tx] = f2bf(tile[tx][r]);
  }
}

// ---------------- positional softmax: pos[p][n] ----------------
__global__ void pos_kernel(const float* __restrict__ coords,
                           const float* __restrict__ pe,
                           float* __restrict__ pos) {
  int p = blockIdx.x, t = threadIdx.x;
  __shared__ float red[256];
  float pev[6];
#pragma unroll
  for (int k = 0; k < 6; ++k) pev[k] = pe[p * 6 + k];
  float l[4];
#pragma unroll
  for (int j = 0; j < 4; ++j) {
    int n = t + 256 * j;
    const float* c = coords + ((size_t)p * NN + n) * 6;
    float s = 0.f;
#pragma unroll
    for (int k = 0; k < 6; ++k) s += c[k] * pev[k];
    l[j] = s;
  }
  float mx = fmaxf(fmaxf(l[0], l[1]), fmaxf(l[2], l[3]));
  red[t] = mx; __syncthreads();
  for (int s = 128; s > 0; s >>= 1) { if (t < s) red[t] = fmaxf(red[t], red[t + s]); __syncthreads(); }
  mx = red[0]; __syncthreads();
  float e[4], sm = 0.f;
#pragma unroll
  for (int j = 0; j < 4; ++j) { e[j] = __expf(l[j] - mx); sm += e[j]; }
  red[t] = sm; __syncthreads();
  for (int s = 128; s > 0; s >>= 1) { if (t < s) red[t] += red[t + s]; __syncthreads(); }
  float inv = 1.0f / red[0];
#pragma unroll
  for (int j = 0; j < 4; ++j) pos[(size_t)p * NN + t + 256 * j] = e[j] * inv;
}

// ---- kb = bf16( Y @ W^T ): one wave per 16 y-rows, loops 48 e-tiles ----
__global__ void __launch_bounds__(32)
kgen_kernel(const float* __restrict__ y, const u16* __restrict__ wb,
            u16* __restrict__ kb) {
  __shared__ __align__(16) u16 s_a[16 * 776];   // 16 y-rows bf16, stride 776 u16
  int lane = threadIdx.x;
  int b = blockIdx.x >> 6, nt = blockIdx.x & 63;
  int n0 = nt * 16;
  int m = lane & 15, hf = lane >> 4;

  // stage 16 y-rows f32 -> bf16 (3072 float4's over 32 lanes)
  for (int t3 = 0; t3 < 96; ++t3) {
    int idx = lane + 32 * t3;
    int rr = idx / 192, cc = idx % 192;
    float4 f = *(const float4*)(y + ((size_t)b * NN + n0 + rr) * DD + cc * 4);
    u32* dst = (u32*)(s_a + rr * 776 + cc * 4);
    dst[0] = pk2(f.x, f.y); dst[1] = pk2(f.z, f.w);
  }
  const char* arow = (const char*)s_a + m * 1552 + hf * 16;

  for (int et = 0; et < DD / 16; ++et) {
    const char* brow = (const char*)(wb + (size_t)(et * 16 + m) * DD) + hf * 16;
    if (et + 1 < DD / 16) __builtin_prefetch(brow + 16 * DD * 2, 0, 0);
    v8f acc = {};
#pragma unroll 4
    for (int c = 0; c < DD / 32; ++c) {
      acc = wmma_bf16(fragA_row(arow, c * 64), fragB_row(brow, c * 64), acc);
    }
    AccF ac; ac.v = acc;
#pragma unroll
    for (int r = 0; r < 8; ++r) {
      kb[((size_t)b * NN + n0 + r + 8 * hf) * DD + et * 16 + m] = f2bf(ac.f[r]);
    }
  }
}

// ---------------- fused attention: one wave per (b, 16-row tile) ----------------
__global__ void __launch_bounds__(32)
attn_kernel(const float* __restrict__ x, const u16* __restrict__ kb,
            const u16* __restrict__ ybT, const float* __restrict__ pos,
            const float* __restrict__ gating, const float* __restrict__ tempp,
            float* __restrict__ out, float* __restrict__ hmap) {
  extern __shared__ u16 smem[];
  u16* s_x    = smem;                 // 16 rows, stride 776 u16 (24832 B)
  u16* s_attn = smem + 16 * 776;      // 16 rows, stride 1032 u16 (33024 B)

  int lane = threadIdx.x;
  int b = blockIdx.x >> 6, rt = blockIdx.x & 63;
  int row0 = rt * 16;
  int m = lane & 15, hf = lane >> 4;

  // phase 0: stage 16 x-rows, f32 -> bf16
  for (int t3 = 0; t3 < 96; ++t3) {
    int idx = lane + 32 * t3;
    int rr = idx / 192, cc = idx % 192;
    float4 f = *(const float4*)(x + ((size_t)b * NN + row0 + rr) * DD + cc * 4);
    u32* dst = (u32*)(s_x + rr * 776 + cc * 4);
    dst[0] = pk2(f.x, f.y); dst[1] = pk2(f.z, f.w);
  }
  const char* xrow = (const char*)s_x + m * 1552 + hf * 16;

  // phase 1: S = x @ k^T; store exp(S*scale) bf16 (logits ~N(0,1): exp-safe)
  for (int j = 0; j < NN / 16; ++j) {
    const char* brow = (const char*)(kb + ((size_t)b * NN + j * 16 + m) * DD) + hf * 16;
    if (j + 1 < NN / 16) __builtin_prefetch(brow + 16 * DD * 2, 0, 0);
    v8f acc = {};
#pragma unroll 4
    for (int c = 0; c < DD / 32; ++c) {
      acc = wmma_bf16(fragA_row(xrow, c * 64), fragB_row(brow, c * 64), acc);
    }
    AccF ac; ac.v = acc;
#pragma unroll
    for (int r = 0; r < 8; ++r) {
      float e = __expf(ac.f[r] * QK_SCALE);
      s_attn[(r + 8 * hf) * 1032 + j * 16 + m] = f2bf(e);
    }
  }

  // softmax normalize + gating mix + entropy (lane: row m, column half hf)
  float gt  = gating[0];
  float g   = 1.0f / (1.0f + __expf(-gt));
  float cg  = 1.0f - g;
  float tmp = tempp[0];
  float sm = 0.f;
  uint4* prow = (uint4*)((char*)s_attn + m * 2064 + hf * 1024);   // 64 uint4 = 512 bf16
  for (int i = 0; i < 64; ++i) sm += sum8(prow[i]);
  sm += __shfl_xor(sm, 16, 32);
  float inv = 1.0f / sm;
  float ent = 0.f;
  const float4* pp = (const float4*)(pos + (size_t)(row0 + m) * NN + hf * 512);
  for (int i = 0; i < 64; ++i) {
    uint4 q = prow[i];
    float4 p0 = pp[2 * i], p1 = pp[2 * i + 1];
    float a0 = cg * bflo(q.x) * inv + g * p0.x;
    float a1 = cg * bfhi(q.x) * inv + g * p0.y;
    float a2 = cg * bflo(q.y) * inv + g * p0.z;
    float a3 = cg * bfhi(q.y) * inv + g * p0.w;
    float a4 = cg * bflo(q.z) * inv + g * p1.x;
    float a5 = cg * bfhi(q.z) * inv + g * p1.y;
    float a6 = cg * bflo(q.w) * inv + g * p1.z;
    float a7 = cg * bfhi(q.w) * inv + g * p1.w;
    ent -= a0 * __logf(a0 + 1e-8f) + a1 * __logf(a1 + 1e-8f) +
           a2 * __logf(a2 + 1e-8f) + a3 * __logf(a3 + 1e-8f) +
           a4 * __logf(a4 + 1e-8f) + a5 * __logf(a5 + 1e-8f) +
           a6 * __logf(a6 + 1e-8f) + a7 * __logf(a7 + 1e-8f);
    q.x = pk2(a0, a1); q.y = pk2(a2, a3); q.z = pk2(a4, a5); q.w = pk2(a6, a7);
    prow[i] = q;
  }
  ent += __shfl_xor(ent, 16, 32);
  if (hf == 0) {
    float hv = 2.0f * (1.0f - 1.0f / (1.0f + __expf(-tmp * ent)));
    hmap[(size_t)b * NN + row0 + m] = hv;
  }

  // phase 2: out = attn @ y   (B operand streams from ybT rows)
  const char* arow = (const char*)s_attn + m * 2064 + hf * 16;
  for (int t = 0; t < DD / 16; ++t) {
    const char* brow = (const char*)(ybT + ((size_t)b * DD + t * 16 + m) * NN) + hf * 16;
    if (t + 1 < DD / 16) __builtin_prefetch(brow + 16 * NN * 2, 0, 0);
    v8f acc = {};
#pragma unroll 4
    for (int c2 = 0; c2 < NN / 32; ++c2) {
      acc = wmma_bf16(fragA_row(arow, c2 * 64), fragB_row(brow, c2 * 64), acc);
    }
    AccF ac; ac.v = acc;
#pragma unroll
    for (int r = 0; r < 8; ++r) {
      out[((size_t)b * NN + row0 + r + 8 * hf) * DD + t * 16 + m] = ac.f[r];
    }
  }
}

extern "C" void kernel_launch(void* const* d_in, const int* in_sizes, int n_in,
                              void* d_out, int out_size, void* d_ws, size_t ws_size,
                              hipStream_t stream) {
  (void)in_sizes; (void)n_in; (void)out_size; (void)ws_size;
  const float* x      = (const float*)d_in[0];
  const float* y      = (const float*)d_in[1];
  const float* coords = (const float*)d_in[2];
  const float* W      = (const float*)d_in[3];
  const float* pe     = (const float*)d_in[4];
  const float* gating = (const float*)d_in[5];
  const float* temp   = (const float*)d_in[6];

  char* ws = (char*)d_ws;
  u16*   kb  = (u16*)ws;                               // 25,165,824 B
  u16*   wb  = (u16*)(ws + 25165824);                  //  1,179,648 B
  float* pos = (float*)(ws + 26345472);                //  4,194,304 B
  u16*   ybT = (u16*)(ws + 30539776);                  // 25,165,824 B  (total ~55.7 MB)

  float* out  = (float*)d_out;
  float* hmap = out + (size_t)BB * NN * DD;

  convw_kernel<<<(DD * DD) / 4 / 256, 256, 0, stream>>>(W, wb);
  pos_kernel<<<NN, 256, 0, stream>>>(coords, pe, pos);
  transp_kernel<<<BB * (NN / 32) * (DD / 32), 256, 0, stream>>>(y, ybT);
  kgen_kernel<<<BB * (NN / 16), 32, 0, stream>>>(y, wb, kb);
  attn_kernel<<<BB * (NN / 16), 32, 57856, stream>>>(x, kb, ybT, pos, gating, temp,
                                                     out, hmap);
}
